// DynConvLayer_14791867367932
// MI455X (gfx1250) — matrix operand order repaired and checked by reading, not compiled, not run on hardware
//
#include <hip/hip_runtime.h>
#include <hip/hip_bf16.h>
#include <math.h>

#define CDIM   64
#define HDIM   512
#define WDIM   512
#define HWSZ   (HDIM*WDIM)
#define TW     64          // output pixels per block (one row segment)
#define TSW    68          // staged tile width = TW + 4 (dilation-2 halo)
#define CPAD   72          // LDS channel pitch (bf16): 144B rows, 16B-aligned
#define NCHUNK 18          // K chunks: 9 taps * 2 halves of 32 channels
#define OPITCH 65          // LDS output transpose pitch (floats)

typedef __attribute__((ext_vector_type(16))) __bf16 v16bf;
typedef __attribute__((ext_vector_type(8)))  __bf16 v8bf;
typedef __attribute__((ext_vector_type(8)))  float  v8f;

__device__ __forceinline__ int refl(int i, int n) {
  i = (i < 0) ? -i : i;
  return (i >= n) ? (2*n - 2 - i) : i;
}
__device__ __forceinline__ float gelu_exact(float x) {
  return 0.5f * x * (1.0f + erff(x * 0.7071067811865476f));
}
__device__ __forceinline__ v16bf cat8(v8bf lo, v8bf hi) {
  return __builtin_shufflevector(lo, hi, 0,1,2,3,4,5,6,7,8,9,10,11,12,13,14,15);
}

// ---- async-stage a 3-row x 68-px x 64-ch reflected tile from the channels-last
//      bf16 image into LDS (16B per async op; ASYNCcnt-tracked, then barrier)
__device__ __forceinline__ void stage_tile_async(const __bf16* __restrict__ srccl,
                                                 int y, int x0, __bf16* tileS) {
  const unsigned int lds_base = (unsigned int)(unsigned long long)tileS;
  const unsigned long long gbase = (unsigned long long)srccl;
  for (int i = threadIdx.x; i < 3*TSW*8; i += 256) {
    int c8 = i & 7;
    int px = (i >> 3) % TSW;
    int r  = (i >> 3) / TSW;             // 0..2 -> ky
    int yy = refl(y + (r - 1)*2, HDIM);
    int xx = refl(x0 - 2 + px, WDIM);
    unsigned int ldsa = lds_base + (unsigned int)(((r*TSW + px)*CPAD + c8*8) * 2);
    unsigned int goff = (unsigned int)(((yy*WDIM + xx)*CDIM + c8*8) * 2);
    asm volatile("global_load_async_to_lds_b128 %0, %1, %2 offset:0"
                 :: "v"(ldsa), "v"(goff), "s"(gbase) : "memory");
  }
  asm volatile("s_wait_asynccnt 0" ::: "memory");
}

// ---- implicit-GEMM conv tile via bf16 WMMA; raw accumulators -> outS[c][px]
__device__ __forceinline__ void wmma_conv(const __bf16* tileS,
                                          const __bf16* __restrict__ wrep,
                                          float* outS) {
  const int lane = threadIdx.x & 31;
  const int w    = threadIdx.x >> 5;   // 0..7 waves
  const int cg   = w & 3;              // c_out group (16 channels)
  const int pg0  = (w >> 2) * 2;       // first of two pixel groups
  const int m    = lane & 15;
  const int sel  = lane >> 4;

  // B fragments (weights): 18 chunks x 16 bf16/lane (compiler may stream them)
  v16bf wb[NCHUNK];
  const v8bf* wp = (const v8bf*)(wrep + (size_t)(cg*32 + lane) * (NCHUNK*16));
  #pragma unroll
  for (int k = 0; k < NCHUNK; ++k) wb[k] = cat8(wp[2*k], wp[2*k+1]);

  #pragma unroll
  for (int pp = 0; pp < 2; ++pp) {
    const int pg = pg0 + pp;
    v8f acc = {0.f,0.f,0.f,0.f,0.f,0.f,0.f,0.f};
    #pragma unroll
    for (int tap = 0; tap < 9; ++tap) {
      const int r  = tap / 3;
      const int dx = (tap % 3) * 2;               // kx offset {-2,0,2} -> {0,2,4}
      const __bf16* tp = tileS + (r*TSW + pg*16 + m + dx) * CPAD + sel*8;
      v8bf a0 = *(const v8bf*)tp;                 // K 0..7 / 8..15
      v8bf a1 = *(const v8bf*)(tp + 16);          // K 16..23 / 24..31
      acc = __builtin_amdgcn_wmma_f32_16x16x32_bf16(
              false, cat8(a0, a1), false, wb[2*tap], (short)0, acc, false, false);
      v8bf b0 = *(const v8bf*)(tp + 32);          // channels 32..63
      v8bf b1 = *(const v8bf*)(tp + 48);
      acc = __builtin_amdgcn_wmma_f32_16x16x32_bf16(
              false, cat8(b0, b1), false, wb[2*tap+1], (short)0, acc, false, false);
    }
    const int co = cg*16 + m;                     // D: row=pixel, col=c_out
    const int pb = pg*16 + 8*sel;
    #pragma unroll
    for (int g = 0; g < 8; ++g) outS[co*OPITCH + pb + g] = acc[g];
  }
}

// ---- repack OIHW fp32 weights into per-lane B-fragment order (bf16)
__global__ void k_repack(const float* __restrict__ w, __bf16* __restrict__ wrep) {
  int i = blockIdx.x * blockDim.x + threadIdx.x;
  if (i >= 4*32*NCHUNK*16) return;
  int j     = i & 15;
  int t     = i >> 4;
  int chunk = t % NCHUNK;
  int t2    = t / NCHUNK;
  int lane  = t2 & 31;
  int cg    = t2 >> 5;
  int tap   = chunk >> 1;
  int half  = chunk & 1;
  int co    = cg*16 + (lane & 15);
  int cin   = half*32 + (j >> 3)*16 + (lane >> 4)*8 + (j & 7);
  wrep[i] = (__bf16)w[((co*CDIM + cin)*3 + tap/3)*3 + (tap % 3)];
}

// ---- NCHW fp32 -> channels-last bf16 image (one-time transpose via LDS)
__global__ __launch_bounds__(256) void k_tocl(const float* __restrict__ x,
                                              __bf16* __restrict__ xcl) {
  __shared__ __align__(16) __bf16 t[TW*CDIM];
  const int y  = blockIdx.x >> 3;
  const int x0 = (blockIdx.x & 7) * TW;
  for (int i = threadIdx.x; i < TW*CDIM; i += 256) {
    int px = i & (TW-1);                 // fastest -> coalesced reads
    int c  = i >> 6;
    t[px*CDIM + c] = (__bf16)x[(size_t)c*HWSZ + y*WDIM + x0 + px];
  }
  __syncthreads();
  for (int i = threadIdx.x; i < TW*8; i += 256) {
    int c8 = i & 7;                      // fastest -> contiguous 16B writes
    int px = i >> 3;
    *(v8bf*)&xcl[((size_t)(y*WDIM + x0 + px))*CDIM + c8*8] =
        *(const v8bf*)&t[px*CDIM + c8*8];
  }
}

// ---- kernel 1: out1 = gelu(conv1(x)+b1); x_ori = where(dilate(mask), out1, x)
//      writes x_ori as channels-last bf16 (conv2's input precision anyway)
__global__ __launch_bounds__(256) void k_conv1(
    const float* __restrict__ x, const __bf16* __restrict__ xcl,
    const float* __restrict__ mask,
    const __bf16* __restrict__ wrep1, const float* __restrict__ b1,
    __bf16* __restrict__ xoricl) {
  __shared__ __align__(16) __bf16 tileS[3*TSW*CPAD];
  __shared__ float outS[CDIM*OPITCH];
  __shared__ float mrowS[TSW];
  __shared__ float mdS[TW];

  const int y  = blockIdx.x >> 3;
  const int x0 = (blockIdx.x & 7) * TW;

  stage_tile_async(xcl, y, x0, tileS);
  if (threadIdx.x < TSW) {                        // 5-row column max of mask
    int col = x0 - 2 + (int)threadIdx.x;
    float mv = 0.0f;
    if (col >= 0 && col < WDIM) {
      for (int dy = -2; dy <= 2; ++dy) {
        int yy = y + dy;
        if (yy >= 0 && yy < HDIM) mv = fmaxf(mv, mask[yy*WDIM + col]);
      }
    }
    mrowS[threadIdx.x] = mv;
  }
  __syncthreads();
  if (threadIdx.x < TW) {                         // 5-wide row max -> dilated mask
    float mv = mrowS[threadIdx.x];
    #pragma unroll
    for (int k = 1; k < 5; ++k) mv = fmaxf(mv, mrowS[threadIdx.x + k]);
    mdS[threadIdx.x] = mv;
  }
  wmma_conv(tileS, wrep1, outS);
  __syncthreads();

  for (int i = threadIdx.x; i < CDIM*TW; i += 256) {   // merge in LDS
    int px = i & (TW-1);
    int c  = i >> 6;
    float xv = x[(size_t)c*HWSZ + y*WDIM + x0 + px];
    float o  = gelu_exact(outS[c*OPITCH + px] + b1[c]);
    outS[c*OPITCH + px] = (mdS[px] > 0.5f) ? o : xv;
  }
  __syncthreads();
  for (int i = threadIdx.x; i < TW*8; i += 256) {      // channels-last bf16 out
    int c8 = i & 7;
    int px = i >> 3;
    v8bf o8;
    #pragma unroll
    for (int j = 0; j < 8; ++j) o8[j] = (__bf16)outS[(c8*8 + j)*OPITCH + px];
    *(v8bf*)&xoricl[((size_t)(y*WDIM + x0 + px))*CDIM + c8*8] = o8;
  }
}

// ---- kernel 2: out2=gelu(conv2(x_ori)+b2); out3=gelu(dwconv(x)+b3);
//      out = where(mask, out2, out3) + x
__global__ __launch_bounds__(256) void k_conv2(
    const float* __restrict__ x, const __bf16* __restrict__ xoricl,
    const float* __restrict__ mask,
    const __bf16* __restrict__ wrep2, const float* __restrict__ b2,
    const float* __restrict__ w3, const float* __restrict__ b3,
    float* __restrict__ out) {
  __shared__ __align__(16) __bf16 tileS[3*TSW*CPAD];
  __shared__ float outS[CDIM*OPITCH];

  const int y  = blockIdx.x >> 3;
  const int x0 = (blockIdx.x & 7) * TW;

  stage_tile_async(xoricl, y, x0, tileS);
  __syncthreads();
  wmma_conv(tileS, wrep2, outS);
  __syncthreads();

  for (int i = threadIdx.x; i < CDIM*TW; i += 256) {
    int px = i & (TW-1);
    int c  = i >> 6;
    int gx = x0 + px;
    size_t gi = (size_t)c*HWSZ + y*WDIM + gx;
    float xv = x[gi];
    float v;
    if (mask[y*WDIM + gx] > 0.5f) {
      v = gelu_exact(outS[c*OPITCH + px] + b2[c]);
    } else {
      float s = b3[c];
      #pragma unroll
      for (int t = 0; t < 9; ++t) {
        int yy = refl(y + (t/3 - 1)*2, HDIM);
        int xx = refl(gx + (t%3 - 1)*2, WDIM);
        s += w3[c*9 + t] * x[(size_t)c*HWSZ + yy*WDIM + xx];
      }
      v = gelu_exact(s);
    }
    out[gi] = v + xv;
  }
}

extern "C" void kernel_launch(void* const* d_in, const int* in_sizes, int n_in,
                              void* d_out, int out_size, void* d_ws, size_t ws_size,
                              hipStream_t stream) {
  const float* x    = (const float*)d_in[0];
  const float* mask = (const float*)d_in[1];
  const float* w1   = (const float*)d_in[2];
  const float* b1   = (const float*)d_in[3];
  const float* w2   = (const float*)d_in[4];
  const float* b2   = (const float*)d_in[5];
  const float* w3   = (const float*)d_in[6];
  const float* b3   = (const float*)d_in[7];
  float* out = (float*)d_out;

  const int wn = 4*32*NCHUNK*16;                   // 36864 bf16 per conv
  __bf16* wrep1  = (__bf16*)d_ws;
  __bf16* wrep2  = wrep1 + wn;
  __bf16* xcl    = (__bf16*)((char*)d_ws + (size_t)2*wn*sizeof(__bf16)); // 256B-aligned
  __bf16* xoricl = xcl + (size_t)HWSZ*CDIM;

  k_repack<<<(wn + 255)/256, 256, 0, stream>>>(w1, wrep1);
  k_repack<<<(wn + 255)/256, 256, 0, stream>>>(w2, wrep2);

  const int nblocks = HDIM * (WDIM / TW);          // 4096
  k_tocl <<<nblocks, 256, 0, stream>>>(x, xcl);
  k_conv1<<<nblocks, 256, 0, stream>>>(x, xcl, mask, wrep1, b1, xoricl);
  k_conv2<<<nblocks, 256, 0, stream>>>(x, xoricl, mask, wrep2, b2, w3, b3, out);
}
